// MambaCipherSolver_9569187136012
// MI455X (gfx1250) — matrix operand order repaired and checked by reading, not compiled, not run on hardware
//
#include <hip/hip_runtime.h>
#include <math.h>

#define D_MODEL   512
#define N_LAYERS  6
#define VOCAB     256
#define D_STATE   128
#define D_CONV    4
#define HEADDIM   64
#define D_INNER   1024
#define NHEADS    16
#define CONV_DIM  1280
#define D_IN_PROJ 2320
#define BATCH     4
#define SEQ       2048
#define NTOK      (BATCH * SEQ)   // 8192 rows

typedef __bf16 bf16x16 __attribute__((ext_vector_type(16)));
typedef float  v8f     __attribute__((ext_vector_type(8)));

union FragAB { uint4 u[2]; bf16x16 v; };

__device__ __forceinline__ unsigned short f2bf(float f) {
  unsigned int u = __float_as_uint(f);
  u += 0x7FFFu + ((u >> 16) & 1u);           // round-to-nearest-even
  return (unsigned short)(u >> 16);
}
__device__ __forceinline__ float siluf(float x) { return x / (1.0f + __expf(-x)); }
__device__ __forceinline__ float softplusf(float x) {
  return (x > 20.0f) ? x : log1pf(__expf(x));
}

// CDNA5 async global->LDS copy (ASYNCcnt path). LDS byte offset = low 32 bits
// of the generic pointer (shared->flat addrspacecast is aperture|offset).
__device__ __forceinline__ void async_ld128(const void* g, void* l) {
  unsigned loff = (unsigned)(uintptr_t)l;
  asm volatile("global_load_async_to_lds_b128 %0, %1, off"
               :: "v"(loff), "v"(g) : "memory");
}
__device__ __forceinline__ void wait_async0() {
  asm volatile("s_wait_asynccnt 0x0" ::: "memory");
}

// ---------------------------------------------------------------- embedding
__global__ __launch_bounds__(128) void embed_kernel(const int* __restrict__ x,
                                                    const float* __restrict__ emb,
                                                    float* __restrict__ h) {
  int row = blockIdx.x;                 // token index 0..8191
  int tok = x[row];
  const float4* src = (const float4*)(emb + (size_t)tok * D_MODEL);
  float4*       dst = (float4*)(h + (size_t)row * D_MODEL);
  dst[threadIdx.x] = src[threadIdx.x];  // 128 threads * float4 = 512 floats
}

// ------------------------------------------------- rmsnorm -> bf16 GEMM A
template <int DD>
__global__ __launch_bounds__(256) void rmsnorm_bf16_kernel(const float* __restrict__ in,
                                                           const float* __restrict__ w,
                                                           unsigned short* __restrict__ out) {
  constexpr int CNT = DD / 256;
  int row = blockIdx.x;
  const float* xr = in + (size_t)row * DD;
  float vals[CNT];
  float s = 0.f;
#pragma unroll
  for (int i = 0; i < CNT; ++i) {
    float v = xr[threadIdx.x + 256 * i];
    vals[i] = v; s += v * v;
  }
#pragma unroll
  for (int o = 16; o >= 1; o >>= 1) s += __shfl_xor(s, o);
  __shared__ float wsum[8];
  __shared__ float scale_s;
  int lane = threadIdx.x & 31, wid = threadIdx.x >> 5;
  if (lane == 0) wsum[wid] = s;
  __syncthreads();
  if (threadIdx.x == 0) {
    float t = 0.f;
#pragma unroll
    for (int i = 0; i < 8; ++i) t += wsum[i];
    scale_s = rsqrtf(t / (float)DD + 1e-5f);
  }
  __syncthreads();
  float sc = scale_s;
#pragma unroll
  for (int i = 0; i < CNT; ++i) {
    int d = threadIdx.x + 256 * i;
    out[(size_t)row * DD + d] = f2bf(vals[i] * sc * w[d]);
  }
}

// ------------------------- gated rmsnorm: g = y*silu(z); rms(g)*gw -> bf16
__global__ __launch_bounds__(256) void gated_rmsnorm_bf16_kernel(const float* __restrict__ y,
                                                                 const float* __restrict__ zx,
                                                                 const float* __restrict__ gw,
                                                                 unsigned short* __restrict__ out) {
  int row = blockIdx.x;
  const float* yr = y + (size_t)row * D_INNER;
  const float* zr = zx + (size_t)row * D_IN_PROJ;   // z = first 1024 cols of zxbcdt
  float vals[4];
  float s = 0.f;
#pragma unroll
  for (int i = 0; i < 4; ++i) {
    int d = threadIdx.x + 256 * i;
    float g = yr[d] * siluf(zr[d]);
    vals[i] = g; s += g * g;
  }
#pragma unroll
  for (int o = 16; o >= 1; o >>= 1) s += __shfl_xor(s, o);
  __shared__ float wsum[8];
  __shared__ float scale_s;
  int lane = threadIdx.x & 31, wid = threadIdx.x >> 5;
  if (lane == 0) wsum[wid] = s;
  __syncthreads();
  if (threadIdx.x == 0) {
    float t = 0.f;
#pragma unroll
    for (int i = 0; i < 8; ++i) t += wsum[i];
    scale_s = rsqrtf(t / (float)D_INNER + 1e-5f);
  }
  __syncthreads();
  float sc = scale_s;
#pragma unroll
  for (int i = 0; i < 4; ++i) {
    int d = threadIdx.x + 256 * i;
    out[(size_t)row * D_INNER + d] = f2bf(vals[i] * sc * gw[d]);
  }
}

// ------------------------------------------- f32 weight -> padded bf16
__global__ __launch_bounds__(256) void convert_w_kernel(const float* __restrict__ w,
                                                        unsigned short* __restrict__ out,
                                                        int N, int Npad, int K) {
  int total = Npad * K;
  for (int idx = blockIdx.x * 256 + threadIdx.x; idx < total; idx += gridDim.x * 256) {
    int n = idx / K;
    float v = (n < N) ? w[idx] : 0.f;   // row-major [Npad][K]; n<N indices coincide
    out[idx] = f2bf(v);
  }
}

// ------------------------------- WMMA GEMM: C[M,N](f32) (+)= A[M,K] * W[N,K]^T
// block tile 128x64, 8 waves, each wave 32x32 (2x2 v_wmma_f32_16x16x32_bf16).
// Tiles staged via async global->LDS (double-buffered, ASYNCcnt-tracked).
__global__ __launch_bounds__(256) void gemm_bf16_wmma(const unsigned short* __restrict__ A,
                                                      const unsigned short* __restrict__ W,
                                                      float* __restrict__ C,
                                                      int N, int K, int add) {
  __shared__ unsigned short lA[2][128 * 48];   // 48-half stride: 96B rows, 16B aligned
  __shared__ unsigned short lW[2][64 * 48];
  const int tid  = threadIdx.x;
  const int lane = tid & 31, wid = tid >> 5;
  const int wm = wid & 3, wn = wid >> 2;        // 4 waves along M, 2 along N
  const int mrow = lane & 15, hi = lane >> 4;
  const int blockM = blockIdx.y * 128;
  const int blockN = blockIdx.x * 64;

  v8f acc[4];
#pragma unroll
  for (int i = 0; i < 4; ++i)
#pragma unroll
    for (int j = 0; j < 8; ++j) acc[i][j] = 0.f;

  const int srow = tid >> 2;      // staging: 64 rows x (4 chunks of 8 halves)
  const int skg  = tid & 3;

  // issue all three 16B async copies this thread owns for tile at k0 -> buf
  auto issue_tile = [&](int k0, int buf) {
    async_ld128(A + (size_t)(blockM + srow) * K + k0 + skg * 8,
                &lA[buf][srow * 48 + skg * 8]);
    async_ld128(A + (size_t)(blockM + srow + 64) * K + k0 + skg * 8,
                &lA[buf][(srow + 64) * 48 + skg * 8]);
    async_ld128(W + (size_t)(blockN + srow) * K + k0 + skg * 8,
                &lW[buf][srow * 48 + skg * 8]);
  };

  issue_tile(0, 0);
  int cur = 0;
  for (int k0 = 0; k0 < K; k0 += 32) {
    wait_async0();        // this wave's async copies for tile k0 landed in LDS
    __syncthreads();      // all waves' copies landed; prior reads of buf done
    if (k0 + 32 < K) issue_tile(k0 + 32, cur ^ 1);   // overlap with compute

    FragAB af[2], bfm[2];
#pragma unroll
    for (int ms = 0; ms < 2; ++ms) {  // A layout: i<8 -> K=8*hi+i ; i>=8 -> K=16+8*hi+(i-8)
      const unsigned short* p = &lA[cur][(wm * 32 + ms * 16 + mrow) * 48 + 8 * hi];
      af[ms].u[0] = *(const uint4*)p;
      af[ms].u[1] = *(const uint4*)(p + 16);
    }
#pragma unroll
    for (int ns = 0; ns < 2; ++ns) {  // B layout: lane col n, i<8 -> K=8*hi+i ; i>=8 -> K=16+8*hi+(i-8)
      const unsigned short* p = &lW[cur][(wn * 32 + ns * 16 + mrow) * 48 + 8 * hi];
      bfm[ns].u[0] = *(const uint4*)p;
      bfm[ns].u[1] = *(const uint4*)(p + 16);
    }
#pragma unroll
    for (int ms = 0; ms < 2; ++ms)
#pragma unroll
      for (int ns = 0; ns < 2; ++ns)
        acc[ms * 2 + ns] = __builtin_amdgcn_wmma_f32_16x16x32_bf16(
            false, af[ms].v, false, bfm[ns].v, (short)0, acc[ms * 2 + ns], false, false);
    cur ^= 1;
  }

  // epilogue: C layout row = j + 8*hi, col = lane&15
#pragma unroll
  for (int ms = 0; ms < 2; ++ms)
#pragma unroll
    for (int ns = 0; ns < 2; ++ns) {
      int col = blockN + wn * 32 + ns * 16 + mrow;
      if (col < N) {
        int rbase = blockM + wm * 32 + ms * 16 + 8 * hi;
        v8f a = acc[ms * 2 + ns];
#pragma unroll
        for (int j = 0; j < 8; ++j) {
          size_t o = (size_t)(rbase + j) * N + col;
          if (add) C[o] += a[j]; else C[o] = a[j];
        }
      }
    }
}

// ------------------------------------ depthwise causal conv (k=4) + silu
__global__ __launch_bounds__(256) void conv_silu_kernel(const float* __restrict__ zx,
                                                        const float* __restrict__ cw,
                                                        const float* __restrict__ cb,
                                                        float* __restrict__ out) {
  const int total = BATCH * SEQ * CONV_DIM;
  for (int idx = blockIdx.x * 256 + threadIdx.x; idx < total; idx += gridDim.x * 256) {
    int c = idx % CONV_DIM;
    int s = (idx / CONV_DIM) % SEQ;
    int b = idx / (CONV_DIM * SEQ);
    float acc = cb[c];
    const float* col = zx + D_INNER + c;   // xBC slice of zxbcdt
    int rowbase = b * SEQ;
#pragma unroll
    for (int j = 0; j < D_CONV; ++j) {
      int sp = s - (D_CONV - 1) + j;
      if (sp >= 0) acc += col[(size_t)(rowbase + sp) * D_IN_PROJ] * cw[c * D_CONV + j];
    }
    out[idx] = siluf(acc);
  }
}

// ------------------------------------------------- selective state scan
// one block per (b, head); 512 threads; 16 state elems/thread (64x128 state)
__global__ __launch_bounds__(512) void scan_kernel(const float* __restrict__ zx,
                                                   const float* __restrict__ xc,
                                                   const float* __restrict__ dtb,
                                                   const float* __restrict__ Alog,
                                                   const float* __restrict__ Dvec,
                                                   float* __restrict__ yout) {
  const int b  = blockIdx.x >> 4;
  const int hd = blockIdx.x & 15;
  __shared__ __align__(16) float xs[HEADDIM];
  __shared__ __align__(16) float Bs[D_STATE];
  __shared__ __align__(16) float Cs[D_STATE];
  __shared__ float sh_dt, sh_decay, sh_A, sh_bias, sh_D;
  if (threadIdx.x == 0) {
    sh_A = -__expf(Alog[hd]);
    sh_bias = dtb[hd];
    sh_D = Dvec[hd];
  }
  __syncthreads();
  const float A = sh_A, bias = sh_bias, Dh = sh_D;

  const int p  = threadIdx.x >> 3;   // head dim 0..63
  const int nc = threadIdx.x & 7;    // state chunk 0..7 (16 n-values each)
  float hstate[16];
#pragma unroll
  for (int i = 0; i < 16; ++i) hstate[i] = 0.f;

  for (int t = 0; t < SEQ; ++t) {
    const size_t rrow = (size_t)(b * SEQ + t);
    const float* xr = xc + rrow * CONV_DIM;
    if (threadIdx.x < 320) {
      int i = threadIdx.x;
      if (i < 64)       xs[i]       = xr[hd * HEADDIM + i];
      else if (i < 192) Bs[i - 64]  = xr[D_INNER + (i - 64)];
      else              Cs[i - 192] = xr[D_INNER + D_STATE + (i - 192)];
    } else if (threadIdx.x == 320) {
      float raw = zx[rrow * D_IN_PROJ + (D_INNER + CONV_DIM) + hd] + bias;
      float dt = softplusf(raw);
      sh_dt = dt;
      sh_decay = __expf(dt * A);
    }
    __syncthreads();
    const float dt = sh_dt, decay = sh_decay;
    const float dtx = dt * xs[p];
    float acc = 0.f;
    const float4* Bp4 = (const float4*)(Bs + nc * 16);
    const float4* Cp4 = (const float4*)(Cs + nc * 16);
#pragma unroll
    for (int j = 0; j < 4; ++j) {
      float4 Bv = Bp4[j], Cv = Cp4[j];
      float* hh = hstate + j * 4;
      hh[0] = hh[0] * decay + dtx * Bv.x; acc += hh[0] * Cv.x;
      hh[1] = hh[1] * decay + dtx * Bv.y; acc += hh[1] * Cv.y;
      hh[2] = hh[2] * decay + dtx * Bv.z; acc += hh[2] * Cv.z;
      hh[3] = hh[3] * decay + dtx * Bv.w; acc += hh[3] * Cv.w;
    }
    acc += __shfl_xor(acc, 1);
    acc += __shfl_xor(acc, 2);
    acc += __shfl_xor(acc, 4);
    if (nc == 0) yout[rrow * D_INNER + hd * HEADDIM + p] = acc + Dh * xs[p];
    __syncthreads();
  }
}

// ================================================================ launch
static inline int cdiv(int a, int b) { return (a + b - 1) / b; }

extern "C" void kernel_launch(void* const* d_in, const int* in_sizes, int n_in,
                              void* d_out, int out_size, void* d_ws, size_t ws_size,
                              hipStream_t stream) {
  const int*   x      = (const int*)d_in[0];
  const float* emb    = (const float*)d_in[1];
  const float* norm_w = (const float*)d_in[2];
  const float* in_w   = (const float*)d_in[3];
  const float* conv_w = (const float*)d_in[4];
  const float* conv_b = (const float*)d_in[5];
  const float* dt_b   = (const float*)d_in[6];
  const float* Alog   = (const float*)d_in[7];
  const float* Dvec   = (const float*)d_in[8];
  const float* gnw    = (const float*)d_in[9];
  const float* ow     = (const float*)d_in[10];
  const float* nfw    = (const float*)d_in[11];
  const float* lmw    = (const float*)d_in[12];
  float* out = (float*)d_out;

  // workspace carve-up
  char* base = (char*)d_ws;
  size_t off = 0;
  auto carve = [&](size_t bytes) { char* p = base + off; off += (bytes + 255) & ~(size_t)255; return p; };
  float*          h    = (float*)carve((size_t)NTOK * D_MODEL * 4);      // residual stream
  unsigned short* A16  = (unsigned short*)carve((size_t)NTOK * D_INNER * 2); // bf16 GEMM A
  unsigned short* W16  = (unsigned short*)carve((size_t)2368 * 512 * 2);     // padded bf16 weights
  float*          zxb  = (float*)carve((size_t)NTOK * D_IN_PROJ * 4);    // zxbcdt
  float*          xcv  = (float*)carve((size_t)NTOK * CONV_DIM * 4);     // conv+silu(xBC)
  float*          ybuf = (float*)carve((size_t)NTOK * D_INNER * 4);      // scan output

  embed_kernel<<<NTOK, 128, 0, stream>>>(x, emb, h);

  for (int l = 0; l < N_LAYERS; ++l) {
    // 1) rmsnorm(h) -> bf16
    rmsnorm_bf16_kernel<D_MODEL><<<NTOK, 256, 0, stream>>>(h, norm_w + l * D_MODEL, A16);
    // 2) in_proj: [8192,512] @ [2320,512]^T -> zxbcdt
    {
      const int N = D_IN_PROJ, K = D_MODEL, Npad = cdiv(N, 64) * 64;
      convert_w_kernel<<<1024, 256, 0, stream>>>(in_w + (size_t)l * N * K, W16, N, Npad, K);
      gemm_bf16_wmma<<<dim3(cdiv(N, 64), NTOK / 128), 256, 0, stream>>>(A16, W16, zxb, N, K, 0);
    }
    // 3) depthwise conv + silu on xBC slice
    conv_silu_kernel<<<cdiv(BATCH * SEQ * CONV_DIM, 256), 256, 0, stream>>>(
        zxb, conv_w + (size_t)l * CONV_DIM * D_CONV, conv_b + (size_t)l * CONV_DIM, xcv);
    // 4) selective scan (softplus(dt) fused), y += D*x
    scan_kernel<<<BATCH * NHEADS, 512, 0, stream>>>(zxb, xcv, dt_b + l * NHEADS,
                                                    Alog + l * NHEADS, Dvec + l * NHEADS, ybuf);
    // 5) gated rmsnorm -> bf16
    gated_rmsnorm_bf16_kernel<<<NTOK, 256, 0, stream>>>(ybuf, zxb, gnw + (size_t)l * D_INNER, A16);
    // 6) out_proj with fused residual add: h += A @ ow^T
    {
      const int N = D_MODEL, K = D_INNER, Npad = cdiv(N, 64) * 64;
      convert_w_kernel<<<1024, 256, 0, stream>>>(ow + (size_t)l * N * K, W16, N, Npad, K);
      gemm_bf16_wmma<<<dim3(cdiv(N, 64), NTOK / 128), 256, 0, stream>>>(A16, W16, h, N, K, 1);
    }
  }

  // final norm + lm_head -> logits (f32, [8192, 256])
  rmsnorm_bf16_kernel<D_MODEL><<<NTOK, 256, 0, stream>>>(h, nfw, A16);
  {
    const int N = VOCAB, K = D_MODEL, Npad = cdiv(N, 64) * 64;
    convert_w_kernel<<<1024, 256, 0, stream>>>(lmw, W16, N, Npad, K);
    gemm_bf16_wmma<<<dim3(cdiv(N, 64), NTOK / 128), 256, 0, stream>>>(A16, W16, out, N, K, 0);
  }
}
